// EfficientViTBlock_2010044694641
// MI455X (gfx1250) — matrix operand order repaired and checked by reading, not compiled, not run on hardware
//
#include <hip/hip_runtime.h>
#include <hip/hip_bf16.h>

// ---------------- types ----------------
typedef __attribute__((ext_vector_type(16))) _Float16 v16h;
typedef __attribute__((ext_vector_type(8)))  _Float16 v8h;
typedef __attribute__((ext_vector_type(8)))  float    v8f;
typedef int v4i __attribute__((vector_size(16)));

// ---------------- problem constants ----------------
#define BB    4
#define CC    1024
#define NN    1024          /* H*W */
#define MIDC  4096          /* C*EXPAND */

// ---------------- workspace layout (bytes) ----------------
static constexpr size_t SZ_WH_QKV  = 3072ULL * 1024 * 2;
static constexpr size_t SZ_WH_PROJ = 1024ULL * 2048 * 2;
static constexpr size_t SZ_WH_INV  = 8192ULL * 1024 * 2;
static constexpr size_t SZ_WH_PT   = 1024ULL * 4096 * 2;
static constexpr size_t SZ_XH      = (size_t)BB * CC * NN * 2;     // [b][n][c] f16
static constexpr size_t SZ_QKV     = (size_t)BB * 3072 * NN * 4;
static constexpr size_t SZ_AGG     = SZ_QKV;
static constexpr size_t SZ_HS      = (size_t)BB * 8192 * NN * 2;   // silu(inv) f16 [c][n]
static constexpr size_t SZ_H2      = (size_t)BB * 4096 * NN * 2;   // gated f16 [c][n]
static constexpr size_t SZ_ATTNH   = (size_t)BB * NN * 2048 * 2;   // [b][n][c] f16
static constexpr size_t SZ_PROJO   = (size_t)BB * CC * NN * 4;
static constexpr size_t SZ_X1      = SZ_PROJO;
static constexpr size_t SZ_X1H     = SZ_XH;                        // [b][n][c] f16

static constexpr size_t O_WH_QKV  = 0;
static constexpr size_t O_WH_PROJ = O_WH_QKV  + SZ_WH_QKV;
static constexpr size_t O_WH_INV  = O_WH_PROJ + SZ_WH_PROJ;
static constexpr size_t O_WH_PT   = O_WH_INV  + SZ_WH_INV;
static constexpr size_t O_XH      = O_WH_PT   + SZ_WH_PT;
static constexpr size_t O_R1      = O_XH      + SZ_XH;       // union region
static constexpr size_t O_QKV     = O_R1;                    // phase 1
static constexpr size_t O_AGG     = O_R1 + SZ_QKV;           // phase 1
static constexpr size_t O_HS      = O_R1;                    // phase 2 (qkv/agg dead)
static constexpr size_t O_H2      = O_R1 + SZ_HS;            // phase 2
static constexpr size_t SZ_R1     = SZ_QKV + SZ_AGG;         // == SZ_HS + SZ_H2
static constexpr size_t O_ATTNH   = O_R1    + SZ_R1;
static constexpr size_t O_PROJO   = O_ATTNH + SZ_ATTNH;
static constexpr size_t O_X1      = O_PROJO + SZ_PROJO;
static constexpr size_t O_X1H     = O_X1    + SZ_X1;
static constexpr size_t O_PTO     = O_X1H   + SZ_X1H;

// ---------------- CDNA5 async global->LDS helpers ----------------
__device__ __forceinline__ void lds_async_b128(const _Float16* g, _Float16* l) {
#if __has_builtin(__builtin_amdgcn_global_load_async_to_lds_b128)
    __builtin_amdgcn_global_load_async_to_lds_b128(
        (__attribute__((address_space(1))) v4i*)(uintptr_t)g,
        (__attribute__((address_space(3))) v4i*)(uintptr_t)l, 0, 0);
#else
    asm volatile("global_load_async_to_lds_b128 %0, %1, off"
                 :: "v"((unsigned)(uintptr_t)l), "v"(g) : "memory");
#endif
}
__device__ __forceinline__ void wait_async0() {
#if __has_builtin(__builtin_amdgcn_s_wait_asynccnt)
    __builtin_amdgcn_s_wait_asynccnt(0);
#else
    asm volatile("s_wait_asynccnt 0" ::: "memory");
#endif
}

// ---------------- f32 -> f16 convert (weights, layout preserved) ----------------
__global__ void k_f32_to_f16(const float* __restrict__ in, _Float16* __restrict__ out, size_t n) {
    size_t i = (size_t)blockIdx.x * blockDim.x + threadIdx.x;
    if (i < n) out[i] = (_Float16)in[i];
}

// ---------------- f32 [b][c][n] -> f16 [b][n][c] transposing convert ----------------
// grid: (NN/32, C/32, B), block 256 (32 x 8)
template<int C>
__global__ void k_transpose_to_f16(const float* __restrict__ in, _Float16* __restrict__ out) {
    __shared__ float tile[32][33];
    const size_t b = blockIdx.z;
    const int n0 = blockIdx.x * 32, c0 = blockIdx.y * 32;
    const int tx = threadIdx.x & 31, ty = threadIdx.x >> 5;     // ty 0..7
    const float* src = in + b * (size_t)C * NN;
    #pragma unroll
    for (int i = 0; i < 32; i += 8)
        tile[ty + i][tx] = src[(size_t)(c0 + ty + i) * NN + n0 + tx];
    __syncthreads();
    _Float16* dst = out + b * (size_t)NN * C;
    #pragma unroll
    for (int i = 0; i < 32; i += 8)
        dst[(size_t)(n0 + ty + i) * C + c0 + tx] = (_Float16)tile[tx][ty + i];
}

// ---------------- WMMA GEMM: Out[b,o,n] = sum_i W[o,i]*Act[b,i,n] ----------------
// W: [M][K] f16.  Act: BT ? [b][n][k] f16 : [b][k][n] f16.
// MODE 0: f32 -> OutF[b][m][n].  MODE 1: silu(acc+bias[m]) -> f16 OutH[b][m][n].
// grid: (N/128, M/64, B), block 256 = 8 waves as 2(M) x 4(N); wave tile 32x32.
__device__ __forceinline__ v8f wmma16(v16h a, v16h b, v8f c) {
    return __builtin_amdgcn_wmma_f32_16x16x32_f16(false, a, false, b, (short)0, c, false, false);
}

template<int M, int N, int K, int MODE, bool BT>
__global__ __launch_bounds__(256)
void k_gemm(const _Float16* __restrict__ Wt, const _Float16* __restrict__ Act,
            float* __restrict__ OutF, _Float16* __restrict__ OutH,
            const float* __restrict__ bias) {
    __shared__ __align__(16) _Float16 As[2][64 * 40];    // [row][k] pad 40
    __shared__ __align__(16) _Float16 Bs[2][128 * 40];   // [n][k]   pad 40

    const int t    = threadIdx.x;
    const int lane = t & 31;
    const int wv   = t >> 5;
    const int wm   = wv & 1;        // 0..1
    const int wn   = wv >> 1;       // 0..3
    const int row0 = blockIdx.y * 64;
    const int n0   = blockIdx.x * 128;
    const size_t b = blockIdx.z;
    const _Float16* ActB = Act + b * (size_t)K * N;   // per-batch size = K*N either layout

    // staging assignments
    const int a_r  = t >> 2;            // 0..63
    const int a_s  = (t & 3) * 8;       // 0,8,16,24
    const _Float16* a_src = Wt + (size_t)(row0 + a_r) * K + a_s;
    const int bt_n = t >> 1;            // 0..127
    const int bt_s = (t & 1) * 16;      // 0,16
    const _Float16* bt_src = ActB + (size_t)(n0 + bt_n) * K + bt_s;
    const int tr_k = t >> 3;            // 0..31
    const int tr_n = (t & 7) * 16;      // 0..112
    const _Float16* tr_src = ActB + (size_t)tr_k * N + n0 + tr_n;

    v8f acc[2][2];
    #pragma unroll
    for (int i = 0; i < 2; ++i)
        #pragma unroll
        for (int j = 0; j < 2; ++j)
            #pragma unroll
            for (int r = 0; r < 8; ++r) acc[i][j][r] = 0.f;

    // ---- prologue: stage chunk 0 (async global->LDS where layout allows) ----
    lds_async_b128(a_src, &As[0][a_r * 40 + a_s]);
    if constexpr (BT) {
        lds_async_b128(bt_src,     &Bs[0][bt_n * 40 + bt_s]);
        lds_async_b128(bt_src + 8, &Bs[0][bt_n * 40 + bt_s + 8]);
    } else {
        v8h q0 = *(const v8h*)tr_src, q1 = *(const v8h*)(tr_src + 8);
        #pragma unroll
        for (int j = 0; j < 8; ++j) { Bs[0][(tr_n + j) * 40 + tr_k] = q0[j];
                                      Bs[0][(tr_n + 8 + j) * 40 + tr_k] = q1[j]; }
    }
    wait_async0();
    __syncthreads();

    const int hs16 = (lane >= 16) ? 1 : 0;
    const int ako  = hs16 * 16;
    const int bko  = hs16 * 16;
    const int lr   = lane & 15;

    constexpr int NCH = K / 32;
    for (int i = 0; i < NCH; ++i) {
        const int cur = i & 1;
        const int nxt = cur ^ 1;
        v8h pb0 = {0,0,0,0,0,0,0,0}, pb1 = {0,0,0,0,0,0,0,0};   // transpose-path regs
        // kick off next chunk (async engine writes LDS while we do WMMA)
        if (i + 1 < NCH) {
            a_src += 32;
            lds_async_b128(a_src, &As[nxt][a_r * 40 + a_s]);
            __builtin_prefetch(a_src + 64, 0, 1);
            if constexpr (BT) {
                bt_src += 32;
                lds_async_b128(bt_src,     &Bs[nxt][bt_n * 40 + bt_s]);
                lds_async_b128(bt_src + 8, &Bs[nxt][bt_n * 40 + bt_s + 8]);
            } else {
                tr_src += (size_t)32 * N;
                pb0 = *(const v8h*)tr_src; pb1 = *(const v8h*)(tr_src + 8);
            }
        }
        // fragments (16-bit A 16x32 / B 32x16; A and B both use contiguous k = 16*(lane/16) + e)
        const _Float16* ap0 = &As[cur][(wm * 32 + lr) * 40];
        const _Float16* ap1 = ap0 + 16 * 40;
        v16h a0, a1;
        #pragma unroll
        for (int e = 0; e < 8; ++e) {
            a0[e] = ap0[ako + e]; a0[e + 8] = ap0[ako + 8 + e];
            a1[e] = ap1[ako + e]; a1[e + 8] = ap1[ako + 8 + e];
        }
        const _Float16* bp0 = &Bs[cur][(wn * 32 + lr) * 40 + bko];
        const _Float16* bp1 = bp0 + 16 * 40;
        v16h b0, b1;
        #pragma unroll
        for (int e = 0; e < 16; ++e) { b0[e] = bp0[e]; b1[e] = bp1[e]; }

        acc[0][0] = wmma16(a0, b0, acc[0][0]);
        acc[0][1] = wmma16(a0, b1, acc[0][1]);
        acc[1][0] = wmma16(a1, b0, acc[1][0]);
        acc[1][1] = wmma16(a1, b1, acc[1][1]);

        // transpose path: scatter next B chunk from registers
        if constexpr (!BT) {
            if (i + 1 < NCH) {
                #pragma unroll
                for (int j = 0; j < 8; ++j) { Bs[nxt][(tr_n + j) * 40 + tr_k] = pb0[j];
                                              Bs[nxt][(tr_n + 8 + j) * 40 + tr_k] = pb1[j]; }
            }
        }
        wait_async0();
        __syncthreads();
    }

    // epilogue: VGPR r -> row r (+8 high lanes), col = lane%16
    const int rl = hs16 * 8;
    #pragma unroll
    for (int am = 0; am < 2; ++am)
        #pragma unroll
        for (int bn = 0; bn < 2; ++bn)
            #pragma unroll
            for (int r = 0; r < 8; ++r) {
                int rowg = row0 + wm * 32 + am * 16 + r + rl;
                int colg = n0 + wn * 32 + bn * 16 + lr;
                size_t idx = b * (size_t)M * N + (size_t)rowg * N + colg;
                float v = acc[am][bn][r];
                if constexpr (MODE == 1) {
                    v += bias[rowg];
                    v = v / (1.f + __expf(-v));
                    OutH[idx] = (_Float16)v;
                } else {
                    OutF[idx] = v;
                }
            }
}

// ---------------- depthwise 5x5, pad 2 (one (b,c) image per block) ----------------
__global__ void k_dwconv5(const float* __restrict__ in, const float* __restrict__ w,
                          float* __restrict__ out) {
    int bc = blockIdx.x;
    int c  = bc % 3072;
    __shared__ float tile[36 * 36];
    __shared__ float wk[25];
    const float* src = in + (size_t)bc * NN;
    int t = threadIdx.x;
    if (t < 25) wk[t] = w[c * 25 + t];
    for (int i = t; i < 36 * 36; i += 256) {
        int y = i / 36 - 2, x = i % 36 - 2;
        float v = 0.f;
        if ((unsigned)y < 32u && (unsigned)x < 32u) v = src[y * 32 + x];
        tile[i] = v;
    }
    __syncthreads();
    for (int p = t; p < 1024; p += 256) {
        int y = p >> 5, x = p & 31;
        float acc = 0.f;
        #pragma unroll
        for (int dy = 0; dy < 5; ++dy)
            #pragma unroll
            for (int dx = 0; dx < 5; ++dx)
                acc += tile[(y + dy) * 36 + x + dx] * wk[dy * 5 + dx];
        out[(size_t)bc * NN + p] = acc;
    }
}

// ---------------- grouped 32x32 pointwise, in place (grid: nchunk,96,B) ----------------
__global__ void k_pw_group(float* __restrict__ data, const float* __restrict__ pw) {
    int g = blockIdx.y, b = blockIdx.z, n0 = blockIdx.x * 64;
    __shared__ float Wl[32 * 32];
    __shared__ float Il[32 * 64];
    int t = threadIdx.x;
    for (int i = t; i < 1024; i += 256) Wl[i] = pw[(size_t)g * 1024 + i];
    float* base = data + ((size_t)b * 3072 + g * 32) * NN + n0;
    for (int i = t; i < 32 * 64; i += 256) {
        int ch = i >> 6, n = i & 63;
        Il[i] = base[(size_t)ch * NN + n];
    }
    __syncthreads();
    for (int i = t; i < 32 * 64; i += 256) {
        int o = i >> 6, n = i & 63;
        float acc = 0.f;
        #pragma unroll
        for (int k = 0; k < 32; ++k) acc += Wl[o * 32 + k] * Il[k * 64 + n];
        base[(size_t)o * NN + n] = acc;
    }
}

// ---------------- linear attention per (head, b); out f16 [b][n][2048] ----------------
__global__ void k_linattn(const float* __restrict__ qkv, const float* __restrict__ agg,
                          _Float16* __restrict__ outh) {
    int h = blockIdx.x, b = blockIdx.y;
    const float* base = (h < 32) ? (qkv + ((size_t)b * 3072 + (size_t)h * 96) * NN)
                                 : (agg + ((size_t)b * 3072 + (size_t)(h - 32) * 96) * NN);
    __shared__ float kbuf[32 * 128];
    __shared__ float vbuf[32 * 128];
    __shared__ float vk[33 * 32];
    int t = threadIdx.x;
    float acc[5] = {0.f, 0.f, 0.f, 0.f, 0.f};
    for (int nc = 0; nc < NN; nc += 128) {
        for (int i = t; i < 32 * 128; i += 256) {
            int r = i >> 7, n = i & 127;
            float kv = base[(size_t)(32 + r) * NN + nc + n];
            kbuf[i] = kv > 0.f ? kv : 0.f;
            vbuf[i] = base[(size_t)(64 + r) * NN + nc + n];
        }
        __syncthreads();
        int pi = 0;
        for (int p = t; p < 1056; p += 256, ++pi) {
            int d = p >> 5, e = p & 31;
            float a = acc[pi];
            if (d == 32) {
                for (int n = 0; n < 128; ++n) a += kbuf[e * 128 + n];          // ones row
            } else {
                for (int n = 0; n < 128; ++n) a += vbuf[d * 128 + n] * kbuf[e * 128 + n];
            }
            acc[pi] = a;
        }
        __syncthreads();
    }
    {
        int pi = 0;
        for (int p = t; p < 1056; p += 256, ++pi) vk[p] = acc[pi];
    }
    __syncthreads();
    for (int n = t; n < NN; n += 256) {
        float q[32];
        #pragma unroll
        for (int e = 0; e < 32; ++e) {
            float v = base[(size_t)e * NN + n];
            q[e] = v > 0.f ? v : 0.f;
        }
        float denom = 0.f;
        #pragma unroll
        for (int e = 0; e < 32; ++e) denom += vk[32 * 32 + e] * q[e];
        denom += 1e-15f;
        _Float16* dst = outh + ((size_t)b * NN + n) * 2048 + (size_t)h * 32;
        for (int d = 0; d < 32; ++d) {
            float a = 0.f;
            #pragma unroll
            for (int e = 0; e < 32; ++e) a += vk[d * 32 + e] * q[e];
            dst[d] = (_Float16)(a / denom);   // contiguous 32-half run per thread
        }
    }
}

// ---------------- RMS over channels + residual; f32 [b][c][n] + optional f16 [b][n][c] ----------------
__global__ void k_rms_add(const float* __restrict__ pre, const float* __restrict__ g,
                          const float* __restrict__ bc, const float* __restrict__ resid,
                          float* __restrict__ outf, _Float16* __restrict__ outh) {
    int b = blockIdx.y;
    int n = blockIdx.x * 256 + threadIdx.x;
    const float* p = pre + (size_t)b * CC * NN;
    float ss = 0.f;
    for (int c = 0; c < CC; ++c) {
        float v = p[(size_t)c * NN + n];
        ss += v * v;
    }
    float inv = rsqrtf(ss * (1.f / CC) + 1e-5f);
    _Float16* oh = outh ? outh + ((size_t)b * NN + n) * CC : nullptr;
    for (int c = 0; c < CC; ++c) {
        size_t idx = (size_t)b * CC * NN + (size_t)c * NN + n;
        float v = p[(size_t)c * NN + n] * inv * g[c] + bc[c] + resid[idx];
        outf[idx] = v;
        if (oh) oh[c] = (_Float16)v;          // contiguous per-thread f16 writes
    }
}

// ---------------- depthwise 3x3 + bias + SwiGLU gate -> f16 [b][c][n] ----------------
__global__ void k_dw3_gate(const _Float16* __restrict__ hs, const float* __restrict__ w,
                           const float* __restrict__ bias, _Float16* __restrict__ out) {
    int bc = blockIdx.x;
    int b = bc >> 12, c = bc & 4095;
    __shared__ float t0[34 * 34];
    __shared__ float t1[34 * 34];
    __shared__ float wk[18];
    const _Float16* s0 = hs + ((size_t)b * 8192 + c) * NN;
    const _Float16* s1 = hs + ((size_t)b * 8192 + c + MIDC) * NN;
    int t = threadIdx.x;
    if (t < 9)       wk[t] = w[(size_t)c * 9 + t];
    else if (t < 18) wk[t] = w[((size_t)c + MIDC) * 9 + (t - 9)];
    for (int i = t; i < 34 * 34; i += 256) {
        int y = i / 34 - 1, x = i % 34 - 1;
        float a = 0.f, bv = 0.f;
        if ((unsigned)y < 32u && (unsigned)x < 32u) { a = (float)s0[y * 32 + x]; bv = (float)s1[y * 32 + x]; }
        t0[i] = a; t1[i] = bv;
    }
    __syncthreads();
    float b0 = bias[c], b1 = bias[c + MIDC];
    for (int p = t; p < 1024; p += 256) {
        int y = p >> 5, x = p & 31;
        float a = b0, gg = b1;
        #pragma unroll
        for (int dy = 0; dy < 3; ++dy)
            #pragma unroll
            for (int dx = 0; dx < 3; ++dx) {
                float in0 = t0[(y + dy) * 34 + x + dx];
                float in1 = t1[(y + dy) * 34 + x + dx];
                a  += in0 * wk[dy * 3 + dx];
                gg += in1 * wk[9 + dy * 3 + dx];
            }
        float sg = gg / (1.f + __expf(-gg));
        out[((size_t)b * MIDC + c) * NN + p] = (_Float16)(a * sg);
    }
}

// ---------------- host-side orchestration ----------------
extern "C" void kernel_launch(void* const* d_in, const int* in_sizes, int n_in,
                              void* d_out, int out_size, void* d_ws, size_t ws_size,
                              hipStream_t stream) {
    (void)in_sizes; (void)n_in; (void)out_size; (void)ws_size;
    const float* x      = (const float*)d_in[0];
    const float* qkv_w  = (const float*)d_in[1];
    const float* agg_dw = (const float*)d_in[2];
    const float* agg_pw = (const float*)d_in[3];
    const float* proj_w = (const float*)d_in[4];
    const float* proj_g = (const float*)d_in[5];
    const float* proj_b = (const float*)d_in[6];
    const float* inv_w  = (const float*)d_in[7];
    const float* inv_b  = (const float*)d_in[8];
    const float* dw_w   = (const float*)d_in[9];
    const float* dw_b   = (const float*)d_in[10];
    const float* pt_w   = (const float*)d_in[11];
    const float* pt_g   = (const float*)d_in[12];
    const float* pt_b   = (const float*)d_in[13];
    float* out = (float*)d_out;
    char*  ws  = (char*)d_ws;

    _Float16* whqkv  = (_Float16*)(ws + O_WH_QKV);
    _Float16* whproj = (_Float16*)(ws + O_WH_PROJ);
    _Float16* whinv  = (_Float16*)(ws + O_WH_INV);
    _Float16* whpt   = (_Float16*)(ws + O_WH_PT);
    _Float16* xh     = (_Float16*)(ws + O_XH);
    float*    qkvF   = (float*)   (ws + O_QKV);
    float*    aggF   = (float*)   (ws + O_AGG);
    _Float16* hsH    = (_Float16*)(ws + O_HS);
    _Float16* h2H    = (_Float16*)(ws + O_H2);
    _Float16* attnH  = (_Float16*)(ws + O_ATTNH);
    float*    projF  = (float*)   (ws + O_PROJO);
    float*    x1F    = (float*)   (ws + O_X1);
    _Float16* x1H    = (_Float16*)(ws + O_X1H);
    float*    ptF    = (float*)   (ws + O_PTO);

    auto cvt = [&](const float* src, _Float16* dst, size_t n) {
        k_f32_to_f16<<<(unsigned)((n + 255) / 256), 256, 0, stream>>>(src, dst, n);
    };
    cvt(qkv_w, whqkv, 3072ULL * 1024);
    cvt(proj_w, whproj, 1024ULL * 2048);
    cvt(inv_w, whinv, 8192ULL * 1024);
    cvt(pt_w, whpt, 1024ULL * 4096);
    // x: f32 [b][c][n] -> f16 [b][n][c]
    k_transpose_to_f16<CC><<<dim3(NN / 32, CC / 32, BB), 256, 0, stream>>>(x, xh);

    // qkv = qkv_w @ x           (B transposed layout, async staging)
    k_gemm<3072, NN, 1024, 0, true><<<dim3(NN / 128, 3072 / 64, BB), 256, 0, stream>>>(
        whqkv, xh, qkvF, nullptr, nullptr);
    // agg pre: depthwise 5x5
    k_dwconv5<<<BB * 3072, 256, 0, stream>>>(qkvF, agg_dw, aggF);
    // grouped pointwise (in place on aggF)
    k_pw_group<<<dim3(NN / 64, 96, BB), 256, 0, stream>>>(aggF, agg_pw);
    // linear attention -> f16 [b][n][2048]
    k_linattn<<<dim3(64, BB), 256, 0, stream>>>(qkvF, aggF, attnH);
    // proj = proj_w @ attn      (B transposed layout, async staging)
    k_gemm<1024, NN, 2048, 0, true><<<dim3(NN / 128, 1024 / 64, BB), 256, 0, stream>>>(
        whproj, attnH, projF, nullptr, nullptr);
    // x1 = x + rms(proj); also f16 [b][n][c] copy for next GEMM
    k_rms_add<<<dim3(NN / 256, BB), 256, 0, stream>>>(projF, proj_g, proj_b, x, x1F, x1H);
    // h = silu(inv_w @ x1 + inv_b) -> f16 [c][n]  (qkv/agg region dead; reuse)
    k_gemm<8192, NN, 1024, 1, true><<<dim3(NN / 128, 8192 / 64, BB), 256, 0, stream>>>(
        whinv, x1H, nullptr, hsH, inv_b);
    // depthwise 3x3 + bias, SwiGLU gate -> f16 [c][n]
    k_dw3_gate<<<BB * MIDC, 256, 0, stream>>>(hsH, dw_w, dw_b, h2H);
    // pt = pt_w @ h2            (B channel-major -> LDS transpose path)
    k_gemm<1024, NN, 4096, 0, false><<<dim3(NN / 128, 1024 / 64, BB), 256, 0, stream>>>(
        whpt, h2H, ptF, nullptr, nullptr);
    // out = x1 + rms(pt)
    k_rms_add<<<dim3(NN / 256, BB), 256, 0, stream>>>(ptF, pt_g, pt_b, x1F, out, nullptr);
}
